// BiWKVRWKV6Encoder_50955491999884
// MI455X (gfx1250) — matrix operand order repaired and checked by reading, not compiled, not run on hardware
//
#include <hip/hip_runtime.h>

// ---------------------------------------------------------------------------
// Problem constants (match reference)
// ---------------------------------------------------------------------------
constexpr int DM  = 1024;           // d_model
constexpr int TS  = 1024;           // T
constexpr int NB  = 8;              // batch
constexpr int HH  = 16;             // heads
constexpr int FF  = 3584;           // d_ffn
constexpr int NT  = NB * TS;        // 8192 rows
constexpr int NTD = NT * DM;        // 8,388,608 elements

// ---------------------------------------------------------------------------
// gfx1250 async global->LDS staging (guarded; falls back to VGPR path)
// ---------------------------------------------------------------------------
#if defined(__has_builtin)
# if __has_builtin(__builtin_amdgcn_global_load_async_to_lds_b128) && \
     __has_builtin(__builtin_amdgcn_s_wait_asynccnt)
#  define USE_ASYNC_LDS 1
# endif
#endif
#ifndef USE_ASYNC_LDS
# define USE_ASYNC_LDS 0
#endif

typedef int v4i_async __attribute__((ext_vector_type(4)));
#define GLOBAL_AS __attribute__((address_space(1)))
#define LDS_AS    __attribute__((address_space(3)))

// ---------------------------------------------------------------------------
// Types for WMMA
// ---------------------------------------------------------------------------
typedef __attribute__((ext_vector_type(16))) __bf16 bf16x16;
typedef __attribute__((ext_vector_type(8)))  __bf16 bf16x8;
typedef __attribute__((ext_vector_type(8)))  float  f32x8;

__device__ __forceinline__ unsigned short f2bf(float f) {
  unsigned int u = __float_as_uint(f);
  unsigned int r = u + 0x7FFFu + ((u >> 16) & 1u);   // round-to-nearest-even
  return (unsigned short)(r >> 16);
}

// ---------------------------------------------------------------------------
// GEMM epilogue modes
// ---------------------------------------------------------------------------
enum {
  EP_F32 = 0,      // C(f32)      = acc
  EP_BF16,         // C(bf16)     = acc
  EP_TANH_BF16,    // C(bf16)     = tanh(acc)
  EP_EXPNEG_F32,   // C(f32)      = exp(-exp(acc + bias[col]))
  EP_RELU2_BF16,   // C(bf16)     = relu(acc)^2
  EP_SILU_F32,     // C(f32)      = acc * sigmoid(acc)
  EP_SIGMOID_F32,  // C(f32)      = sigmoid(acc)
  EP_ADD_F32,      // C(f32)     += acc
  EP_MULADD_F32    // C(f32)     += acc * mul[idx]
};

__device__ __forceinline__ void epilogue(void* C, int ldc, int r, int c, float acc,
                                         int mode, const float* bias, const float* mul) {
  size_t idx = (size_t)r * ldc + c;
  switch (mode) {
    case EP_F32:        ((float*)C)[idx] = acc; break;
    case EP_BF16:       ((unsigned short*)C)[idx] = f2bf(acc); break;
    case EP_TANH_BF16:  ((unsigned short*)C)[idx] = f2bf(tanhf(acc)); break;
    case EP_EXPNEG_F32: ((float*)C)[idx] = expf(-expf(acc + bias[c])); break;
    case EP_RELU2_BF16: { float t = fmaxf(acc, 0.f); ((unsigned short*)C)[idx] = f2bf(t * t); } break;
    case EP_SILU_F32:   ((float*)C)[idx] = acc / (1.f + expf(-acc)); break;
    case EP_SIGMOID_F32:((float*)C)[idx] = 1.f / (1.f + expf(-acc)); break;
    case EP_ADD_F32:    ((float*)C)[idx] += acc; break;
    case EP_MULADD_F32: ((float*)C)[idx] += acc * mul[idx]; break;
  }
}

// ---------------------------------------------------------------------------
// bf16 WMMA GEMM: C(MxN) = A(MxK, bf16 row-major, lda) * B(KxN, bf16 row-major, ldb)
// Block tile 128x64, K-step 32. 8 waves arranged 4(M) x 2(N); each wave owns a
// 32x32 output tile = 2x2 accumulators -> 4 v_wmma per K-step with A/B
// fragment reuse. A tile staged via async global->LDS (gfx1250 ASYNCcnt path)
// when available. Requires: M % 128 == 0, K % 32 == 0 (N may be ragged).
// ---------------------------------------------------------------------------
#define BMX 128
#define BNX 64
#define BKX 32
#define AS_STRIDE 40   // elements; 80B rows keep 16B alignment for frag loads
#define BT_STRIDE 48   // elements; 96B rows keep 32B alignment for frag loads

__global__ __launch_bounds__(256)
void gemm_bf16_wmma(const unsigned short* __restrict__ A, int lda,
                    const unsigned short* __restrict__ Bm, int ldb,
                    void* __restrict__ C, int ldc,
                    int M, int N, int K,
                    int mode, const float* __restrict__ bias,
                    const float* __restrict__ mul)
{
  __shared__ __align__(32) unsigned short As[BMX * AS_STRIDE];
  __shared__ __align__(32) unsigned short Bt[BNX * BT_STRIDE];

  const int tid  = threadIdx.x;
  const int m0   = blockIdx.y * BMX;
  const int n0   = blockIdx.x * BNX;

  const int lane = tid & 31;
  const int wid  = tid >> 5;
  const int hf   = lane >> 4;        // lane half (0/1)
  const int lm   = lane & 15;
  const int wm   = (wid & 3) * 32;   // wave M base (two 16-row tiles)
  const int wn   = (wid >> 2) * 32;  // wave N base (two 16-col tiles)

  f32x8 acc00 = {0.f,0.f,0.f,0.f,0.f,0.f,0.f,0.f};
  f32x8 acc01 = acc00, acc10 = acc00, acc11 = acc00;

  const int brow = tid >> 3;         // 0..31 (k within tile)
  const int bcol = (tid & 7) * 8;    // 0..56 (n within tile)
  const bool nfull = (n0 + BNX) <= N;

  for (int k0 = 0; k0 < K; k0 += BKX) {
    // ---- stage A tile (128x32), 2 x 16B per thread ----
#pragma unroll
    for (int it = 0; it < 2; ++it) {
      const int s  = tid + it * 256;
      const int ar = s >> 2;            // 0..127
      const int ac = (s & 3) * 8;       // 0,8,16,24
      const unsigned short* ap = A + (size_t)(m0 + ar) * lda + (k0 + ac);
#if USE_ASYNC_LDS
      __builtin_amdgcn_global_load_async_to_lds_b128(
          (v4i_async GLOBAL_AS*)ap,
          (v4i_async LDS_AS*)&As[ar * AS_STRIDE + ac],
          0, 0);
#else
      *(uint4*)&As[ar * AS_STRIDE + ac] = *(const uint4*)ap;
#endif
      if (k0 + BKX < K)                 // gfx1250 global_prefetch_b8
        __builtin_prefetch(ap + BKX, 0, 1);
    }

    // ---- stage B tile transposed: Bt[n][k] = B[k][n] ----
    {
      const unsigned short* bp = Bm + (size_t)(k0 + brow) * ldb;
      const int cbase = n0 + bcol;
      unsigned short tmp[8];
      if (nfull) {
        *(uint4*)tmp = *(const uint4*)(bp + cbase);
      } else {
#pragma unroll
        for (int i = 0; i < 8; ++i) {           // branchless clamp + select
          int cc = cbase + i;
          int cs = (cc < N) ? cc : (N - 1);
          unsigned short v = bp[cs];
          tmp[i] = (cc < N) ? v : (unsigned short)0;
        }
      }
#pragma unroll
      for (int i = 0; i < 8; ++i)
        Bt[(bcol + i) * BT_STRIDE + brow] = tmp[i];
    }
#if USE_ASYNC_LDS
    __builtin_amdgcn_s_wait_asynccnt(0);
#endif
    __syncthreads();

    // ---- A fragments (contiguous k order, same as B):
    // lanes 0-15:  K 0..15; lanes 16-31: K 16..31
    const int akoff = hf * 16;
    bf16x8 a0lo = *reinterpret_cast<const bf16x8*>(&As[(wm + lm) * AS_STRIDE + akoff]);
    bf16x8 a0hi = *reinterpret_cast<const bf16x8*>(&As[(wm + lm) * AS_STRIDE + akoff + 8]);
    bf16x8 a1lo = *reinterpret_cast<const bf16x8*>(&As[(wm + 16 + lm) * AS_STRIDE + akoff]);
    bf16x8 a1hi = *reinterpret_cast<const bf16x8*>(&As[(wm + 16 + lm) * AS_STRIDE + akoff + 8]);
    bf16x16 af0 = __builtin_shufflevector(a0lo, a0hi, 0,1,2,3,4,5,6,7,8,9,10,11,12,13,14,15);
    bf16x16 af1 = __builtin_shufflevector(a1lo, a1hi, 0,1,2,3,4,5,6,7,8,9,10,11,12,13,14,15);

    // ---- B fragments (contiguous k order, same as A):
    // lanes 0-15 hold K 0..15 of column lm; lanes 16-31 hold K 16..31
    const int bkoff = hf * 16;
    bf16x16 bf0 = *reinterpret_cast<const bf16x16*>(&Bt[(wn + lm) * BT_STRIDE + bkoff]);
    bf16x16 bf1 = *reinterpret_cast<const bf16x16*>(&Bt[(wn + 16 + lm) * BT_STRIDE + bkoff]);

    acc00 = __builtin_amdgcn_wmma_f32_16x16x32_bf16(false, af0, false, bf0, (short)0, acc00, false, false);
    acc01 = __builtin_amdgcn_wmma_f32_16x16x32_bf16(false, af0, false, bf1, (short)0, acc01, false, false);
    acc10 = __builtin_amdgcn_wmma_f32_16x16x32_bf16(false, af1, false, bf0, (short)0, acc10, false, false);
    acc11 = __builtin_amdgcn_wmma_f32_16x16x32_bf16(false, af1, false, bf1, (short)0, acc11, false, false);
    __syncthreads();
  }

  // ---- store: C/D layout: VGPR i -> row M=i (lanes 0-15) / M=i+8 (lanes 16-31)
#pragma unroll
  for (int i = 0; i < 8; ++i) {
    int r0 = m0 + wm + i + hf * 8;
    int r1 = r0 + 16;
    int c0 = n0 + wn + lm;
    int c1 = c0 + 16;
    if (c0 < N) {
      epilogue(C, ldc, r0, c0, acc00[i], mode, bias, mul);
      epilogue(C, ldc, r1, c0, acc10[i], mode, bias, mul);
    }
    if (c1 < N) {
      epilogue(C, ldc, r0, c1, acc01[i], mode, bias, mul);
      epilogue(C, ldc, r1, c1, acc11[i], mode, bias, mul);
    }
  }
}

// ---------------------------------------------------------------------------
// f32 -> bf16 conversion (weights)
// ---------------------------------------------------------------------------
__global__ void cvt_bf16_kernel(const float* __restrict__ in,
                                unsigned short* __restrict__ out, int n) {
  int i = blockIdx.x * 256 + threadIdx.x;
  if (i < n) out[i] = f2bf(in[i]);
}

// ---------------------------------------------------------------------------
// Elementwise kernels (all over NTD, exact grids)
// ---------------------------------------------------------------------------
__global__ void add_pe_kernel(float* __restrict__ out, const float* __restrict__ x) {
  int idx = blockIdx.x * 256 + threadIdx.x;
  int col = idx & (DM - 1);
  int row = idx >> 10;
  int t   = row & (TS - 1);
  int i2  = col >> 1;
  float div = expf((float)(2 * i2) * (-9.210340371976184f / (float)DM)); // -ln(1e4)/D
  float ang = (float)t * div;
  float pe  = (col & 1) ? cosf(ang) : sinf(ang);
  out[idx] = x[idx] + pe;
}

__global__ void shift_diff_kernel(float* __restrict__ xx, const float* __restrict__ xln) {
  int idx = blockIdx.x * 256 + threadIdx.x;
  int row = idx >> 10;
  int t   = row & (TS - 1);
  float prev = (t > 0) ? xln[idx - DM] : 0.f;
  xx[idx] = prev - xln[idx];
}

// out_bf16 = bf16( x + xx * (vec[col] + dyn[idx]) ), dyn optional
__global__ void mixbf_kernel(unsigned short* __restrict__ out,
                             const float* __restrict__ x, const float* __restrict__ xx,
                             const float* __restrict__ vec, const float* __restrict__ dyn) {
  int idx = blockIdx.x * 256 + threadIdx.x;
  int col = idx & (DM - 1);
  float m = vec[col] + (dyn ? dyn[idx] : 0.f);
  out[idx] = f2bf(x[idx] + xx[idx] * m);
}

__global__ void reverse_kernel(float* __restrict__ out, const float* __restrict__ in) {
  int idx = blockIdx.x * 256 + threadIdx.x;
  int col = idx & (DM - 1);
  int row = idx >> 10;
  int b = row >> 10, t = row & (TS - 1);
  out[idx] = in[(((size_t)(b << 10) + (TS - 1 - t)) << 10) + col];
}

__global__ void combine_kernel(float* __restrict__ out, const float* __restrict__ yf,
                               const float* __restrict__ yb) {
  int idx = blockIdx.x * 256 + threadIdx.x;
  int col = idx & (DM - 1);
  int row = idx >> 10;
  int b = row >> 10, t = row & (TS - 1);
  float rb = yb[(((size_t)(b << 10) + (TS - 1 - t)) << 10) + col];
  out[idx] = 0.5f * (yf[idx] + rb);
}

__global__ void mask_kernel(float* __restrict__ out, const float* __restrict__ in,
                            const int* __restrict__ len) {
  int idx = blockIdx.x * 256 + threadIdx.x;
  int row = idx >> 10;
  int b = row >> 10, t = row & (TS - 1);
  out[idx] = (t < len[b]) ? in[idx] : 0.f;
}

// ---------------------------------------------------------------------------
// LayerNorm over last dim (D=1024), one row per block (256 threads x 4 elems)
// ---------------------------------------------------------------------------
__global__ __launch_bounds__(256)
void layernorm_kernel(float* __restrict__ out, const float* __restrict__ in,
                      const float* __restrict__ w, const float* __restrict__ b,
                      float eps) {
  const int row = blockIdx.x;
  const float* x = in + (size_t)row * DM;
  float* y = out + (size_t)row * DM;
  const int tid = threadIdx.x;

  float s = 0.f, s2 = 0.f;
#pragma unroll
  for (int i = 0; i < 4; ++i) {
    float v = x[tid + i * 256];
    s += v; s2 += v * v;
  }
  __shared__ float a1[256], a2[256];
  a1[tid] = s; a2[tid] = s2;
  __syncthreads();
  for (int off = 128; off > 0; off >>= 1) {
    if (tid < off) { a1[tid] += a1[tid + off]; a2[tid] += a2[tid + off]; }
    __syncthreads();
  }
  float mean = a1[0] * (1.f / DM);
  float var  = a2[0] * (1.f / DM) - mean * mean;
  float inv  = rsqrtf(var + eps);
#pragma unroll
  for (int i = 0; i < 4; ++i) {
    int c = tid + i * 256;
    y[c] = (x[c] - mean) * inv * w[c] + b[c];
  }
}

// ---------------------------------------------------------------------------
// GroupNorm per head (64 ch) fused with gate multiply, bf16 out.
// grid = NT*HH blocks of 64 threads.
// ---------------------------------------------------------------------------
__global__ __launch_bounds__(64)
void gnmul_kernel(unsigned short* __restrict__ outbf, const float* __restrict__ o,
                  const float* __restrict__ g, const float* __restrict__ w,
                  const float* __restrict__ b) {
  const int rowh = blockIdx.x;
  const int row = rowh >> 4, h = rowh & 15;
  const size_t base = (size_t)row * DM + h * 64;
  const int tid = threadIdx.x;

  float v = o[base + tid];
  __shared__ float a1[64], a2[64];
  a1[tid] = v; a2[tid] = v * v;
  __syncthreads();
  for (int off = 32; off > 0; off >>= 1) {
    if (tid < off) { a1[tid] += a1[tid + off]; a2[tid] += a2[tid + off]; }
    __syncthreads();
  }
  float mean = a1[0] * (1.f / 64.f);
  float var  = a2[0] * (1.f / 64.f) - mean * mean;
  float inv  = rsqrtf(var + 64e-5f);          // RWKV-6 ln_x eps
  int c = h * 64 + tid;
  float val = ((v - mean) * inv * w[c] + b[c]) * g[base + tid];
  outbf[base + tid] = f2bf(val);
}

// ---------------------------------------------------------------------------
// WKV6 scan: one (b,h) per block; 256 threads: thread owns V column j = tid&63
// and 16 K-rows of the 64x64 state in registers. Sequential over T.
// Layout of r/k/v/w/o: (B,T,H,64) contiguous == (NT, DM) with head offset.
// ---------------------------------------------------------------------------
__global__ __launch_bounds__(256)
void wkv6_kernel(const float* __restrict__ r, const float* __restrict__ k,
                 const float* __restrict__ v, const float* __restrict__ w,
                 const float* __restrict__ u, float* __restrict__ o) {
  const int bh = blockIdx.x;
  const int b = bh >> 4, h = bh & 15;
  const int tid = threadIdx.x;
  const int j = tid & 63;
  const int kr0 = (tid >> 6) * 16;

  __shared__ float rs[64], ks[64], vs[64], ws[64], us[64];
  __shared__ float red[256];

  if (tid < 64) us[tid] = u[h * 64 + tid];
  float st[16];
#pragma unroll
  for (int i = 0; i < 16; ++i) st[i] = 0.f;

  const size_t base = (size_t)b * TS * DM + h * 64;
  for (int t = 0; t < TS; ++t) {
    size_t idx = base + (size_t)t * DM;
    __syncthreads();
    if (tid < 64)        rs[tid]       = r[idx + tid];
    else if (tid < 128)  ks[tid - 64]  = k[idx + tid - 64];
    else if (tid < 192)  vs[tid - 128] = v[idx + tid - 128];
    else                 ws[tid - 192] = w[idx + tid - 192];
    __syncthreads();

    float vj = vs[j];
    float acc = 0.f;
#pragma unroll
    for (int i = 0; i < 16; ++i) {
      int kk = kr0 + i;
      float kv = ks[kk] * vj;
      acc += rs[kk] * (st[i] + us[kk] * kv);   // uses state BEFORE update
      st[i] = st[i] * ws[kk] + kv;
    }
    red[tid] = acc;
    __syncthreads();
    if (tid < 64)
      o[idx + tid] = red[tid] + red[64 + tid] + red[128 + tid] + red[192 + tid];
  }
}

// ---------------------------------------------------------------------------
// Host-side orchestration
// ---------------------------------------------------------------------------
enum {
  P_LN0W = 2, P_LN0B, P_LN1W, P_LN1B, P_LN2W, P_LN2B,
  P_MAAX, P_MAAW, P_MAAK, P_MAAV, P_MAAR, P_MAAG,
  P_MAAW1, P_MAAW2, P_TDECAY, P_DW1, P_DW2, P_FAAAA,
  P_WR, P_WK, P_WV, P_WG, P_WO, P_LNXW, P_LNXB,
  P_CMAAK, P_CMAAR, P_WCK, P_WCV, P_WCR   // = 31
};

static inline int ceil_div(int a, int b) { return (a + b - 1) / b; }

static void gemm(hipStream_t s, const unsigned short* A, int lda,
                 const unsigned short* B, int ldb, void* C, int ldc,
                 int M, int N, int K, int mode,
                 const float* bias = nullptr, const float* mul = nullptr) {
  dim3 g(ceil_div(N, BNX), ceil_div(M, BMX));
  gemm_bf16_wmma<<<g, 256, 0, s>>>(A, lda, B, ldb, C, ldc, M, N, K, mode, bias, mul);
}

static void cvt(hipStream_t s, unsigned short* dst, const float* src, int n) {
  cvt_bf16_kernel<<<ceil_div(n, 256), 256, 0, s>>>(src, dst, n);
}

struct Ctx {
  const float* P[32];
  float *xln, *xx, *t0, *rb, *kb, *vb, *wb;
  unsigned short *abf, *d1bf, *actbf, *ffnbf, *wslot;
  hipStream_t st;
};

static void run_block(const Ctx& c, int bi, float* xw) {
  hipStream_t s = c.st;
  const int EW = NTD / 256;

  // ================= time mix =================
  layernorm_kernel<<<NT, 256, 0, s>>>(c.xln, xw, c.P[P_LN1W] + bi * DM, c.P[P_LN1B] + bi * DM, 1e-5f);
  shift_diff_kernel<<<EW, 256, 0, s>>>(c.xx, c.xln);

  // a = tanh((x + xx*maa_x) @ maa_w1)  -> bf16 (NT x 160)
  mixbf_kernel<<<EW, 256, 0, s>>>(c.actbf, c.xln, c.xx, c.P[P_MAAX] + bi * DM, nullptr);
  cvt(s, c.wslot, c.P[P_MAAW1] + (size_t)bi * DM * 160, DM * 160);
  gemm(s, c.actbf, DM, c.wslot, 160, c.abf, 160, NT, 160, DM, EP_TANH_BF16);

  const int vecIdx[5] = {P_MAAW, P_MAAK, P_MAAV, P_MAAR, P_MAAG};
  float* gout[5] = {nullptr, c.kb, c.vb, c.rb, nullptr};
  const int wIdx[5] = {-1, P_WK, P_WV, P_WR, -1};
  float* gbuf = c.xx;   // reuse xx for gate after the last mix

  for (int f = 0; f < 5; ++f) {
    // m_f = a[:, f*32:(f+1)*32] @ maa_w2[f]  (NT x DM, f32)
    cvt(s, c.wslot, c.P[P_MAAW2] + ((size_t)bi * 5 + f) * 32 * DM, 32 * DM);
    gemm(s, c.abf + f * 32, 160, c.wslot, DM, c.t0, DM, NT, DM, 32, EP_F32);
    // x_f = x + xx*(maa_vec + m_f)  -> bf16
    mixbf_kernel<<<EW, 256, 0, s>>>(c.actbf, c.xln, c.xx, c.P[vecIdx[f]] + bi * DM, c.t0);

    if (f == 0) {
      // decay LoRA: w = exp(-exp(time_decay + tanh(xw@dw1)@dw2))
      cvt(s, c.wslot, c.P[P_DW1] + (size_t)bi * DM * 64, DM * 64);
      gemm(s, c.actbf, DM, c.wslot, 64, c.d1bf, 64, NT, 64, DM, EP_TANH_BF16);
      cvt(s, c.wslot, c.P[P_DW2] + (size_t)bi * 64 * DM, 64 * DM);
      gemm(s, c.d1bf, 64, c.wslot, DM, c.wb, DM, NT, DM, 64, EP_EXPNEG_F32,
           c.P[P_TDECAY] + bi * DM);
    } else if (f == 4) {
      // g = silu(xg @ Wg)
      cvt(s, c.wslot, c.P[P_WG] + (size_t)bi * DM * DM, DM * DM);
      gemm(s, c.actbf, DM, c.wslot, DM, gbuf, DM, NT, DM, DM, EP_SILU_F32);
    } else {
      cvt(s, c.wslot, c.P[wIdx[f]] + (size_t)bi * DM * DM, DM * DM);
      gemm(s, c.actbf, DM, c.wslot, DM, gout[f], DM, NT, DM, DM, EP_F32);
    }
  }

  // WKV6 scan  (o reuses xln; xln is dead until channel-mix LN)
  float* obuf = c.xln;
  wkv6_kernel<<<NB * HH, 256, 0, s>>>(c.rb, c.kb, c.vb, c.wb,
                                      c.P[P_FAAAA] + (size_t)bi * HH * 64, obuf);
  // GroupNorm per head * g -> bf16
  gnmul_kernel<<<NT * HH, 64, 0, s>>>(c.actbf, obuf, gbuf,
                                      c.P[P_LNXW] + bi * DM, c.P[P_LNXB] + bi * DM);
  // x += (GN(o)*g) @ Wo
  cvt(s, c.wslot, c.P[P_WO] + (size_t)bi * DM * DM, DM * DM);
  gemm(s, c.actbf, DM, c.wslot, DM, xw, DM, NT, DM, DM, EP_ADD_F32);

  // ================= channel mix =================
  layernorm_kernel<<<NT, 256, 0, s>>>(c.xln, xw, c.P[P_LN2W] + bi * DM, c.P[P_LN2B] + bi * DM, 1e-5f);
  shift_diff_kernel<<<EW, 256, 0, s>>>(c.xx, c.xln);

  // kk = relu(xk @ Wck)^2  -> bf16 (NT x FF)
  mixbf_kernel<<<EW, 256, 0, s>>>(c.actbf, c.xln, c.xx, c.P[P_CMAAK] + bi * DM, nullptr);
  cvt(s, c.wslot, c.P[P_WCK] + (size_t)bi * DM * FF, DM * FF);
  gemm(s, c.actbf, DM, c.wslot, FF, c.ffnbf, FF, NT, FF, DM, EP_RELU2_BF16);

  // s = sigmoid(xr @ Wcr) -> t0
  mixbf_kernel<<<EW, 256, 0, s>>>(c.actbf, c.xln, c.xx, c.P[P_CMAAR] + bi * DM, nullptr);
  cvt(s, c.wslot, c.P[P_WCR] + (size_t)bi * DM * DM, DM * DM);
  gemm(s, c.actbf, DM, c.wslot, DM, c.t0, DM, NT, DM, DM, EP_SIGMOID_F32);

  // x += s * (kk @ Wcv)
  cvt(s, c.wslot, c.P[P_WCV] + (size_t)bi * FF * DM, FF * DM);
  gemm(s, c.ffnbf, FF, c.wslot, DM, xw, DM, NT, DM, FF, EP_MULADD_F32, nullptr, c.t0);
}

extern "C" void kernel_launch(void* const* d_in, const int* in_sizes, int n_in,
                              void* d_out, int out_size, void* d_ws, size_t ws_size,
                              hipStream_t stream) {
  (void)in_sizes; (void)n_in; (void)out_size; (void)ws_size;

  const float* x_in  = (const float*)d_in[0];
  const int* lengths = (const int*)d_in[1];

  Ctx c{};
  c.st = stream;
  for (int i = 2; i <= P_WCR; ++i) c.P[i] = (const float*)d_in[i];

  // ---- workspace carve (~403 MB) ----
  char* p = (char*)d_ws;
  auto alloc = [&](size_t bytes) -> void* {
    void* r = (void*)p;
    p += (bytes + 255) & ~(size_t)255;
    return r;
  };
  float* xbuf = (float*)alloc((size_t)NTD * 4);
  float* xwA  = (float*)alloc((size_t)NTD * 4);
  float* xwB  = (float*)alloc((size_t)NTD * 4);
  c.xln = (float*)alloc((size_t)NTD * 4);
  c.xx  = (float*)alloc((size_t)NTD * 4);
  c.t0  = (float*)alloc((size_t)NTD * 4);
  c.rb  = (float*)alloc((size_t)NTD * 4);
  c.kb  = (float*)alloc((size_t)NTD * 4);
  c.vb  = (float*)alloc((size_t)NTD * 4);
  c.wb  = (float*)alloc((size_t)NTD * 4);
  c.abf   = (unsigned short*)alloc((size_t)NT * 160 * 2);
  c.d1bf  = (unsigned short*)alloc((size_t)NT * 64 * 2);
  c.actbf = (unsigned short*)alloc((size_t)NT * DM * 2);
  c.ffnbf = (unsigned short*)alloc((size_t)NT * FF * 2);
  c.wslot = (unsigned short*)alloc((size_t)DM * FF * 2);

  const int EW = NTD / 256;

  // x = x + sinusoidal PE
  add_pe_kernel<<<EW, 256, 0, stream>>>(xbuf, x_in);

  for (int layer = 0; layer < 2; ++layer) {
    // ---- forward direction (layer 0 gets pre-LN ln0) ----
    if (layer == 0) {
      layernorm_kernel<<<NT, 256, 0, stream>>>(xwA, xbuf, c.P[P_LN0W], c.P[P_LN0B], 1e-5f);
    } else {
      (void)hipMemcpyAsync(xwA, xbuf, (size_t)NTD * 4, hipMemcpyDeviceToDevice, stream);
    }
    run_block(c, 2 * layer, xwA);

    // ---- backward direction on time-reversed input ----
    reverse_kernel<<<EW, 256, 0, stream>>>(xwB, xbuf);
    run_block(c, 2 * layer + 1, xwB);

    // ---- x = 0.5*(yf + reverse(yb)) ----
    combine_kernel<<<EW, 256, 0, stream>>>(xbuf, xwA, xwB);
  }

  // masked output
  mask_kernel<<<EW, 256, 0, stream>>>((float*)d_out, xbuf, lengths);
}
